// WABlock_16518444220721
// MI455X (gfx1250) — matrix operand order repaired and checked by reading, not compiled, not run on hardware
//
#include <hip/hip_runtime.h>
#include <hip/hip_bf16.h>

// ---------------------------------------------------------------------------
// Types for WMMA bf16 (gfx1250 / CDNA5, wave32)
// ---------------------------------------------------------------------------
typedef __attribute__((ext_vector_type(16))) __bf16          bf16x16;
typedef __attribute__((ext_vector_type(8)))  float           floatx8;
typedef __attribute__((ext_vector_type(8)))  unsigned short  ushort8;

union Frag16 {
    bf16x16 v;
    ushort8 h[2];
};

__device__ __forceinline__ floatx8 zero8() {
    floatx8 z = {0.f, 0.f, 0.f, 0.f, 0.f, 0.f, 0.f, 0.f};
    return z;
}

// A fragment (16x32 bf16): lane holds row m; elems 0..7 = K k0..k0+7,
// elems 8..15 = K k0+16..k0+23 (lanes 16..31 shifted by +8 in K).
// p = base + row*ld + k0 + (lane>=16 ? 8 : 0)
__device__ __forceinline__ Frag16 load_a(const unsigned short* p) {
    Frag16 f;
    f.h[0] = *(const ushort8*)p;
    f.h[1] = *(const ushort8*)(p + 16);
    return f;
}

// B fragment (32x16 bf16): lane holds column n; elems 0..7 = K k0..k0+7,
// elems 8..15 = K k0+16..k0+23 (lanes 16..31 shifted by +8 in K).
// p = base + col*ld + k0 + (lane>=16 ? 8 : 0)
__device__ __forceinline__ Frag16 load_b(const unsigned short* p) {
    Frag16 f;
    f.h[0] = *(const ushort8*)p;
    f.h[1] = *(const ushort8*)(p + 16);
    return f;
}

__device__ __forceinline__ floatx8 wmma_bf16(const Frag16& a, const Frag16& b, floatx8 c) {
    return __builtin_amdgcn_wmma_f32_16x16x32_bf16(
        /*neg_a=*/false, a.v, /*neg_b=*/false, b.v,
        /*c_mod=*/(short)0, c, /*reuse_a=*/false, /*reuse_b=*/false);
}

__device__ __forceinline__ unsigned short f2bf(float f) {
    unsigned u = __float_as_uint(f);
    unsigned r = u + 0x7FFFu + ((u >> 16) & 1u);   // round-to-nearest-even
    return (unsigned short)(r >> 16);
}

// ---------------------------------------------------------------------------
// Problem constants
// ---------------------------------------------------------------------------
#define BATCH   4
#define CHN     96
#define HH      384
#define WW      384
#define WS_     8
#define NTOK    64      // tokens per window
#define NWIN    48      // windows per axis
#define SHIFT_  4

// Unified q/k/vT LDS pool offsets (in ushort elements)
#define QOFF    0                  // q: h*4096 + m*64 + d   (d padded to 64)
#define KOFF    8192               // k: h*4096 + m*64 + d   (d padded to 64)
#define VOFF    16384              // vT: h*3072 + d*64 + m
#define QKV_ELEMS (16384 + 6144)   // 22528 ushorts = 45 KB

// Workspace layout (bytes)
#define OFF_WQKV   0                               // 288*96 bf16 = 55296 B
#define OFF_WOUT   55296                           // 96*96  bf16 = 18432 B
#define OFF_WC1    73728                           // 96*96  bf16 = 18432 B
#define OFF_WC2    92160                           // 96*864 bf16 = 165888 B
#define OFF_MASK   258048                          // 64*64  f32  = 16384 B
#define OFF_G      274432                          // B*C*H*W bf16 = 113246208 B

// ---------------------------------------------------------------------------
// Prep: weights fp32 -> bf16 ((N,K) row-major), conv2 re-layout, rel-pos mask.
// ---------------------------------------------------------------------------
__global__ void prep_kernel(const float* __restrict__ ipw,
                            const float* __restrict__ opw,
                            const float* __restrict__ c1w,
                            const float* __restrict__ c2w,
                            const float* __restrict__ btab,
                            unsigned short* __restrict__ Wqkv,
                            unsigned short* __restrict__ Wout,
                            unsigned short* __restrict__ Wc1,
                            unsigned short* __restrict__ Wc2,
                            float* __restrict__ mask) {
    int i = blockIdx.x * 256 + threadIdx.x;
    if (i < 288 * 96) Wqkv[i] = f2bf(ipw[i]);
    if (i < 96 * 96) {
        Wout[i] = f2bf(opw[i]);
        Wc1[i]  = f2bf(c1w[i]);
    }
    if (i < 96 * 864) {                       // Wc2[o][tap][c] = conv2_w[o][c][dy][dx]
        int o = i / 864, rem = i % 864;
        int tap = rem / 96, c = rem % 96;
        int dy = tap / 3, dx = tap % 3;
        Wc2[i] = f2bf(c2w[((o * 96 + c) * 3 + dy) * 3 + dx]);
    }
    if (i < NTOK * NTOK) {                    // Swin relative position bias
        int qi = i / NTOK, kj = i % NTOK;
        int yi = qi >> 3, xi = qi & 7, yj = kj >> 3, xj = kj & 7;
        mask[i] = btab[(yi - yj + WS_ - 1) * (2 * WS_ - 1) + (xi - xj + WS_ - 1)];
    }
}

// ---------------------------------------------------------------------------
// Fused shifted-window MHA. 1 block = 1 window, 128 threads = 4 waves,
// wave w owns M-tile w (rows w*16..w*16+15).
// ---------------------------------------------------------------------------
__global__ __launch_bounds__(128) void attn_kernel(
        const float* __restrict__ x,
        const unsigned short* __restrict__ Wqkv,
        const float* __restrict__ bqkv,
        const unsigned short* __restrict__ Wout,
        const float* __restrict__ bout,
        const float* __restrict__ mask,
        float* __restrict__ y) {
    // LDS: 12 KB tile (later O) + 45 KB unified q/k/vT pool (P overlays q)
    __shared__ __align__(16) unsigned short ldsA[NTOK * CHN];
    __shared__ __align__(16) unsigned short ldsU[QKV_ELEMS];

    const int tid   = threadIdx.x;
    const int lane  = tid & 31;
    const int wave  = tid >> 5;           // M-tile
    const int khalf = lane >> 4;          // 0 | 1
    const int col16 = lane & 15;
    const int win   = blockIdx.x;
    const int b     = blockIdx.y;
    const int wy    = win / NWIN, wx = win % NWIN;
    const int row   = wave * 16 + col16;

    // zero head-dim pad (d = 48..63) of q and k
    for (int i = tid; i < 2 * NTOK * 16; i += 128) {
        int h = i / (NTOK * 16), r = (i / 16) % NTOK, d = 48 + (i % 16);
        ldsU[QOFF + h * 4096 + r * 64 + d] = 0;
        ldsU[KOFF + h * 4096 + r * 64 + d] = 0;
    }

    // load shifted window -> bf16 [token][channel]
    for (int i = tid; i < NTOK * CHN; i += 128) {
        int c = i / NTOK, tok = i % NTOK;
        int ty = tok >> 3, tx = tok & 7;
        int gh = (wy * WS_ + ty + SHIFT_) % HH;
        int gw = (wx * WS_ + tx + SHIFT_) % WW;
        ldsA[tok * CHN + c] = f2bf(x[(((size_t)b * CHN + c) * HH + gh) * WW + gw]);
    }
    __syncthreads();

    // ---- qkv GEMM: (64 x 96) @ (96 x 288), branch-free scatter to q/k/vT ----
#pragma unroll 1
    for (int nt = 0; nt < 18; ++nt) {
        floatx8 acc = zero8();
#pragma unroll
        for (int ks = 0; ks < 3; ++ks) {
            Frag16 a = load_a(&ldsA[row * CHN + ks * 32 + khalf * 8]);
            Frag16 bb = load_b(&Wqkv[(nt * 16 + col16) * CHN + ks * 32 + khalf * 8]);
            acc = wmma_bf16(a, bb, acc);
        }
        int ncol  = nt * 16 + col16;         // 0..287
        float bias = bqkv[ncol];
        int which = ncol / CHN;              // 0:q 1:k 2:v  (uniform per wave-iter)
        int cc = ncol % CHN;
        int h = cc / 48, d = cc % 48;
        bool isv = (which == 2);
        // base offset + per-row stride, selected without branches
        int off0    = isv ? (VOFF + h * 3072 + d * 64)
                          : (which * 8192 + h * 4096 + d);
        int strideM = isv ? 1 : 64;
        float scl   = (which == 0) ? 0.14433756729740643f : 1.0f;  // 1/sqrt(48)
#pragma unroll
        for (int r = 0; r < 8; ++r) {
            int m = wave * 16 + r + khalf * 8;
            ldsU[off0 + m * strideM] = f2bf((acc[r] + bias) * scl);
        }
    }
    __syncthreads();

    // ---- scores = q @ k^T + mask (register-resident) ----
    floatx8 sc[2][4];
#pragma unroll
    for (int h = 0; h < 2; ++h) {
#pragma unroll
        for (int nt = 0; nt < 4; ++nt) {
            floatx8 acc = zero8();
#pragma unroll
            for (int ks = 0; ks < 2; ++ks) {
                Frag16 a = load_a(&ldsU[QOFF + h * 4096 + row * 64 + ks * 32 + khalf * 8]);
                Frag16 bb = load_b(&ldsU[KOFF + h * 4096 + (nt * 16 + col16) * 64 +
                                         ks * 32 + khalf * 8]);
                acc = wmma_bf16(a, bb, acc);
            }
            int ncol = nt * 16 + col16;
#pragma unroll
            for (int r = 0; r < 8; ++r) {
                int m = wave * 16 + r + khalf * 8;
                acc[r] += mask[m * NTOK + ncol];
            }
            sc[h][nt] = acc;
        }
    }
    __syncthreads();   // all q reads done; P overlays q region

    // ---- softmax per row (row lives in one 16-lane half), write P ----
#pragma unroll 1
    for (int h = 0; h < 2; ++h) {
#pragma unroll 1
        for (int r = 0; r < 8; ++r) {
            float mx = -1e30f;
#pragma unroll
            for (int nt = 0; nt < 4; ++nt) mx = fmaxf(mx, sc[h][nt][r]);
            for (int off = 8; off >= 1; off >>= 1)
                mx = fmaxf(mx, __shfl_xor(mx, off, 32));
            float e[4], sum = 0.f;
#pragma unroll
            for (int nt = 0; nt < 4; ++nt) {
                e[nt] = __expf(sc[h][nt][r] - mx);
                sum += e[nt];
            }
            for (int off = 8; off >= 1; off >>= 1)
                sum += __shfl_xor(sum, off, 32);
            float inv = 1.0f / sum;
            int m = wave * 16 + r + khalf * 8;
#pragma unroll
            for (int nt = 0; nt < 4; ++nt)
                ldsU[QOFF + h * 4096 + m * 64 + nt * 16 + col16] = f2bf(e[nt] * inv);
        }
    }
    __syncthreads();

    // ---- O = P @ V (K = 64 tokens), O bf16 overlays ldsA ----
#pragma unroll
    for (int h = 0; h < 2; ++h) {
#pragma unroll
        for (int nt = 0; nt < 3; ++nt) {
            floatx8 acc = zero8();
#pragma unroll
            for (int ks = 0; ks < 2; ++ks) {
                Frag16 a = load_a(&ldsU[QOFF + h * 4096 + row * 64 + ks * 32 + khalf * 8]);
                Frag16 bb = load_b(&ldsU[VOFF + h * 3072 + (nt * 16 + col16) * 64 +
                                         ks * 32 + khalf * 8]);
                acc = wmma_bf16(a, bb, acc);
            }
            int d = nt * 16 + col16;
#pragma unroll
            for (int r = 0; r < 8; ++r) {
                int m = wave * 16 + r + khalf * 8;
                ldsA[m * CHN + h * 48 + d] = f2bf(acc[r]);
            }
        }
    }
    __syncthreads();

    // ---- out-proj + bias + residual: y = x + o ----
#pragma unroll 1
    for (int nt = 0; nt < 6; ++nt) {
        floatx8 acc = zero8();
#pragma unroll
        for (int ks = 0; ks < 3; ++ks) {
            Frag16 a = load_a(&ldsA[row * CHN + ks * 32 + khalf * 8]);
            Frag16 bb = load_b(&Wout[(nt * 16 + col16) * CHN + ks * 32 + khalf * 8]);
            acc = wmma_bf16(a, bb, acc);
        }
        int ncol = nt * 16 + col16;
        float bo = bout[ncol];
#pragma unroll
        for (int r = 0; r < 8; ++r) {
            int m = wave * 16 + r + khalf * 8;
            int ty = m >> 3, tx = m & 7;
            int gh = (wy * WS_ + ty + SHIFT_) % HH;
            int gw = (wx * WS_ + tx + SHIFT_) % WW;
            size_t idx = (((size_t)b * CHN + ncol) * HH + gh) * WW + gw;
            y[idx] = x[idx] + acc[r] + bo;
        }
    }
}

// ---------------------------------------------------------------------------
// conv1 (1x1) + exact GELU -> bf16 workspace. 1 block = 64 pixels of one row.
// ---------------------------------------------------------------------------
__global__ __launch_bounds__(128) void conv1_kernel(
        const float* __restrict__ y,
        const unsigned short* __restrict__ Wc1,
        const float* __restrict__ bc1,
        unsigned short* __restrict__ g) {
    __shared__ __align__(16) unsigned short ldsA[64 * CHN];   // [pixel][c]
    const int tid = threadIdx.x, lane = tid & 31, wave = tid >> 5;
    const int khalf = lane >> 4, col16 = lane & 15;
    const int bw = blockIdx.x, b = blockIdx.y;
    const int h = bw / 6, w0 = (bw % 6) * 64;

    for (int i = tid; i < 64 * CHN; i += 128) {
        int c = i / 64, m = i % 64;
        ldsA[m * CHN + c] = f2bf(y[(((size_t)b * CHN + c) * HH + h) * WW + w0 + m]);
    }
    __syncthreads();

    const int row = wave * 16 + col16;
#pragma unroll 1
    for (int nt = 0; nt < 6; ++nt) {
        floatx8 acc = zero8();
#pragma unroll
        for (int ks = 0; ks < 3; ++ks) {
            Frag16 a = load_a(&ldsA[row * CHN + ks * 32 + khalf * 8]);
            Frag16 bb = load_b(&Wc1[(nt * 16 + col16) * CHN + ks * 32 + khalf * 8]);
            acc = wmma_bf16(a, bb, acc);
        }
        int o = nt * 16 + col16;
        float bias = bc1[o];
#pragma unroll
        for (int r = 0; r < 8; ++r) {
            int m = wave * 16 + r + khalf * 8;
            float v = acc[r] + bias;
            float gl = 0.5f * v * (1.0f + erff(v * 0.70710678118654752f));   // exact GELU
            g[(((size_t)b * CHN + o) * HH + h) * WW + w0 + m] = f2bf(gl);
        }
    }
}

// ---------------------------------------------------------------------------
// conv2 (3x3, edge-replicate) + LeakyReLU(0.1) + residual, in-place on y.
// Implicit im2col: K = 9 taps * 96 channels = 864; tap loop fully unrolled
// -> 27 consecutive WMMA k-steps per output tile.
// ---------------------------------------------------------------------------
__global__ __launch_bounds__(128) void conv2_kernel(
        const unsigned short* __restrict__ g,
        const unsigned short* __restrict__ Wc2,
        const float* __restrict__ bc2,
        float* __restrict__ out) {
    __shared__ __align__(16) unsigned short ldsG[3][66][CHN];   // [dy][w(-1..64)][c], 38 KB
    const int tid = threadIdx.x, lane = tid & 31, wave = tid >> 5;
    const int khalf = lane >> 4, col16 = lane & 15;
    const int bw = blockIdx.x, b = blockIdx.y;
    const int h = bw / 6, w0 = (bw % 6) * 64;

    for (int i = tid; i < 3 * 66 * CHN; i += 128) {
        int c = i / 198, r2 = i % 198;
        int dy = r2 / 66, wi = r2 % 66;
        int gh = h + dy - 1;  gh = gh < 0 ? 0 : (gh > HH - 1 ? HH - 1 : gh);   // edge pad
        int gw = w0 + wi - 1; gw = gw < 0 ? 0 : (gw > WW - 1 ? WW - 1 : gw);
        ldsG[dy][wi][c] = g[(((size_t)b * CHN + c) * HH + gh) * WW + gw];
    }
    __syncthreads();

    const int row = wave * 16 + col16;   // local pixel index in [0,64)
#pragma unroll 1
    for (int nt = 0; nt < 6; ++nt) {
        floatx8 acc = zero8();
        const unsigned short* wb = &Wc2[(nt * 16 + col16) * 864 + khalf * 8];
#pragma unroll
        for (int tap = 0; tap < 9; ++tap) {
            const int dy = tap / 3, dx = tap % 3;
#pragma unroll
            for (int ks = 0; ks < 3; ++ks) {
                Frag16 a = load_a(&ldsG[dy][row + dx][ks * 32 + khalf * 8]);
                Frag16 bb = load_b(wb + tap * 96 + ks * 32);
                acc = wmma_bf16(a, bb, acc);
            }
        }
        int o = nt * 16 + col16;
        float bias = bc2[o];
#pragma unroll
        for (int r = 0; r < 8; ++r) {
            int m = wave * 16 + r + khalf * 8;
            size_t idx = (((size_t)b * CHN + o) * HH + h) * WW + w0 + m;
            float v = acc[r] + bias;
            v = v >= 0.f ? v : 0.1f * v;          // LeakyReLU(0.1)
            out[idx] = out[idx] + v;              // += residual (out holds y)
        }
    }
}

// ---------------------------------------------------------------------------
extern "C" void kernel_launch(void* const* d_in, const int* in_sizes, int n_in,
                              void* d_out, int out_size, void* d_ws, size_t ws_size,
                              hipStream_t stream) {
    const float* x    = (const float*)d_in[0];
    const float* ipw  = (const float*)d_in[1];
    const float* ipb  = (const float*)d_in[2];
    const float* opw  = (const float*)d_in[3];
    const float* opb  = (const float*)d_in[4];
    const float* c1w  = (const float*)d_in[5];
    const float* c1b  = (const float*)d_in[6];
    const float* c2w  = (const float*)d_in[7];
    const float* c2b  = (const float*)d_in[8];
    const float* btab = (const float*)d_in[9];

    float* y = (float*)d_out;
    char* ws = (char*)d_ws;
    unsigned short* Wqkv = (unsigned short*)(ws + OFF_WQKV);
    unsigned short* Wout = (unsigned short*)(ws + OFF_WOUT);
    unsigned short* Wc1  = (unsigned short*)(ws + OFF_WC1);
    unsigned short* Wc2  = (unsigned short*)(ws + OFF_WC2);
    float*          msk  = (float*)(ws + OFF_MASK);
    unsigned short* g    = (unsigned short*)(ws + OFF_G);

    prep_kernel<<<324, 256, 0, stream>>>(ipw, opw, c1w, c2w, btab,
                                         Wqkv, Wout, Wc1, Wc2, msk);

    attn_kernel<<<dim3(NWIN * NWIN, BATCH), 128, 0, stream>>>(
        x, Wqkv, ipb, Wout, opb, msk, y);

    conv1_kernel<<<dim3(HH * (WW / 64), BATCH), 128, 0, stream>>>(y, Wc1, c1b, g);

    conv2_kernel<<<dim3(HH * (WW / 64), BATCH), 128, 0, stream>>>(g, Wc2, c2b, y);
}
